// TensorizedCircuit_24910810317487
// MI455X (gfx1250) — matrix operand order repaired and checked by reading, not compiled, not run on hardware
//
#include <hip/hip_runtime.h>
#include <hip/hip_bf16.h>
#include <stdint.h>

typedef __attribute__((ext_vector_type(16))) __bf16 v16bf;
typedef __attribute__((ext_vector_type(8)))  float  v8f;

// ---------------- geometry ----------------
#define KDIM 256
#define BDIM 256
// LDS layout (bf16 strides padded to avoid 64-bank conflicts: 136*2=272B -> 4-bank step)
#define LDA   136
#define LDB   136
#define LDST  264
#define A_BYTES     (256 * LDA * 2)               // 69632
#define B_BYTES     (256 * LDB * 2)               // 69632
#define PART_OFF    (A_BYTES + B_BYTES)           // 139264
#define RECIP_OFF   (PART_OFF + 512 * 4)          // 141312
#define STAGE_OFF   (RECIP_OFF + 256 * 4)         // 142336
#define STAGE_BYTES (256 * LDST * 2)              // 135168
#define SMEM_BYTES  (STAGE_OFF + STAGE_BYTES)     // 277504  (< 320KB WGP LDS)

// ---------------- prologue: h0 = tanh(x[b,d] * W_in[d,k]) as bf16 ----------------
__global__ __launch_bounds__(256) void k_init(const float* __restrict__ x,
                                              const float* __restrict__ Wi,
                                              uint16_t* __restrict__ h0) {
  const int bd = blockIdx.x;          // b*256 + d
  const int d  = bd & 255;
  const float s = __builtin_nontemporal_load(&x[bd]);           // x is (B,D,1)
  const float w = __builtin_nontemporal_load(&Wi[d * 256 + threadIdx.x]);
  const float v = tanhf(s * w);
  reinterpret_cast<__bf16*>(h0)[(size_t)bd * 256 + threadIdx.x] = (__bf16)v;
}

// ---------------- one tree level: per pair p, two 256x256x256 bf16 WMMA GEMMs.
// 512 threads = 16 waves (4/SIMD -> ~256 VGPR budget, no spills), each wave owns a
// 64x64 output tile (acc[4][4]). Softmax fused as exp-on-fill + post-GEMM column
// scaling; arm-0 parked in LDS as bf16; arms multiplied; coalesced bf16 store. ----------------
__global__ __launch_bounds__(512, 1) void k_level(const uint16_t* __restrict__ h_in,
                                                  const float* __restrict__ W,
                                                  uint16_t* __restrict__ h_out) {
  __shared__ __align__(16) unsigned char smem[SMEM_BYTES];
  __bf16* ldsA  = (__bf16*)(smem);               // [row b][k_local], stride LDA
  __bf16* ldsB  = (__bf16*)(smem + A_BYTES);     // [col j][k_local] (transposed), stride LDB
  float*  part  = (float*)(smem + PART_OFF);     // 512 partial colsums
  float*  recip = (float*)(smem + RECIP_OFF);    // 256 reciprocals
  __bf16* stage = (__bf16*)(smem + STAGE_OFF);   // proj0 park + final output staging

  const unsigned* hin_u  = (const unsigned*)h_in;
  unsigned*       hout_u = (unsigned*)h_out;

  const int p      = blockIdx.x;
  const int n_next = gridDim.x;
  const int n_cur  = n_next * 2;
  const int tid    = threadIdx.x;
  const int lane   = tid & 31;
  const int wave   = tid >> 5;    // 0..15
  const int wm     = wave >> 2;   // 0..3 row-wave  (64 output rows)
  const int wn     = wave & 3;    // 0..3 col-wave  (64 output cols)
  const int lr     = lane & 15;
  const bool hihalf = (lane >= 16);

  const int jfill = tid & 255;   // fixed output-column per thread in B fill
  const int gfill = tid >> 8;    // 0..1

  #pragma unroll
  for (int a = 0; a < 2; ++a) {
    const int dcol = 2 * p + a;
    float csum = 0.0f;

    v8f acc[4][4];
    #pragma unroll
    for (int mt = 0; mt < 4; ++mt)
      #pragma unroll
      for (int nt = 0; nt < 4; ++nt)
        #pragma unroll
        for (int c = 0; c < 8; ++c) acc[mt][nt][c] = 0.0f;

    #pragma unroll
    for (int kh = 0; kh < 2; ++kh) {
      __syncthreads();  // previous users of ldsA/ldsB done

      // ---- fill A: h rows (256 x 128 bf16) as 32-bit copies, coalesced ----
      #pragma unroll 8
      for (int i = 0; i < 32; ++i) {
        const int eu = tid + 512 * i;
        const int b  = eu >> 6;
        const int k2 = eu & 63;
        const unsigned v = __builtin_nontemporal_load(
            &hin_u[(size_t)(b * n_cur + dcol) * 128 + kh * 64 + k2]);
        *reinterpret_cast<unsigned*>(&ldsA[b * LDA + k2 * 2]) = v;
      }
      // ---- fill B transposed with exp(W), accumulate column sums ----
      #pragma unroll 8
      for (int i = 0; i < 64; ++i) {
        const int kl = gfill + 2 * i;                       // 0..127
        const float w = __builtin_nontemporal_load(
            &W[((size_t)dcol * 256 + kh * 128 + kl) * 256 + jfill]);
        const float e = __expf(w);
        csum += e;
        ldsB[jfill * LDB + kl] = (__bf16)e;
      }
      __syncthreads();

      // ---- 4 k-steps of bf16 WMMA over this 128-wide half ----
      #pragma unroll
      for (int ks = 0; ks < 4; ++ks) {
        v16bf af[4], bfr[4];
        const int koffA = ks * 32 + (hihalf ? 8 : 0);
        #pragma unroll
        for (int mt = 0; mt < 4; ++mt) {
          const int row = wm * 64 + mt * 16 + lr;
          const __bf16* pa = &ldsA[row * LDA + koffA];
          reinterpret_cast<uint4*>(&af[mt])[0] = *reinterpret_cast<const uint4*>(pa);
          reinterpret_cast<uint4*>(&af[mt])[1] = *reinterpret_cast<const uint4*>(pa + 16);
        }
        const int koffB = ks * 32 + (hihalf ? 8 : 0);
        #pragma unroll
        for (int nt = 0; nt < 4; ++nt) {
          const int col = wn * 64 + nt * 16 + lr;
          const __bf16* pb = &ldsB[col * LDB + koffB];
          reinterpret_cast<uint4*>(&bfr[nt])[0] = *reinterpret_cast<const uint4*>(pb);
          reinterpret_cast<uint4*>(&bfr[nt])[1] = *reinterpret_cast<const uint4*>(pb + 16);
        }
        #pragma unroll
        for (int mt = 0; mt < 4; ++mt)
          #pragma unroll
          for (int nt = 0; nt < 4; ++nt)
            acc[mt][nt] = __builtin_amdgcn_wmma_f32_16x16x32_bf16(
                false, af[mt], false, bfr[nt], (short)0, acc[mt][nt],
                false, false);
      }
    }

    // ---- softmax normalization: scale accumulators by 1/colsum[j] ----
    part[tid] = csum;
    __syncthreads();
    if (tid < 256) {
      const float s = part[tid] + part[tid + 256];
      recip[tid] = 1.0f / s;
    }
    __syncthreads();

    // ---- scale; arm 0: park in LDS; arm 1: multiply with parked arm 0 ----
    #pragma unroll
    for (int mt = 0; mt < 4; ++mt)
      #pragma unroll
      for (int nt = 0; nt < 4; ++nt) {
        const int j = wn * 64 + nt * 16 + lr;
        const float rj = recip[j];
        #pragma unroll
        for (int c = 0; c < 8; ++c) {
          const int row = wm * 64 + mt * 16 + (hihalf ? 8 + c : c);
          const float v = acc[mt][nt][c] * rj;
          if (a == 0) {
            stage[row * LDST + j] = (__bf16)v;            // park proj0 (same lane reads later)
          } else {
            const float p0 = (float)stage[row * LDST + j];
            stage[row * LDST + j] = (__bf16)(p0 * v);     // final h value
          }
        }
      }
  }

  // ---- coalesced global store of staged bf16 output ----
  __syncthreads();
  #pragma unroll 8
  for (int i = 0; i < 32; ++i) {
    const int eu = tid + 512 * i;
    const int b  = eu >> 6;
    const int k2 = eu & 63;
    const unsigned v = *reinterpret_cast<const unsigned*>(&stage[b * LDST + k2 * 2]);
    hout_u[(size_t)(b * n_next + p) * 128 + k2] = v;
  }
}

// ---------------- epilogue: out = root @ softmax(W_out, axis=0) ----------------
__global__ __launch_bounds__(256) void k_out(const uint16_t* __restrict__ rootu,
                                             const float* __restrict__ W_out,
                                             float* __restrict__ out) {
  __shared__ float ldsE[256 * 10];
  __shared__ float rc[10];
  const int tid = threadIdx.x;
  for (int i = tid; i < 2560; i += 256) ldsE[i] = __expf(W_out[i]);
  __syncthreads();
  if (tid < 10) {
    float s = 0.0f;
    for (int k = 0; k < 256; ++k) s += ldsE[k * 10 + tid];
    rc[tid] = 1.0f / s;
  }
  __syncthreads();
  const __bf16* root = (const __bf16*)rootu;
  float acc[10];
  #pragma unroll
  for (int c = 0; c < 10; ++c) acc[c] = 0.0f;
  for (int k = 0; k < 256; ++k) {
    const float r = (float)root[tid * 256 + k];
    #pragma unroll
    for (int c = 0; c < 10; ++c) acc[c] += r * ldsE[k * 10 + c];
  }
  #pragma unroll
  for (int c = 0; c < 10; ++c) out[tid * 10 + c] = acc[c] * rc[c];
}

extern "C" void kernel_launch(void* const* d_in, const int* in_sizes, int n_in,
                              void* d_out, int out_size, void* d_ws, size_t ws_size,
                              hipStream_t stream) {
  const float* x     = (const float*)d_in[0];
  const float* W_in  = (const float*)d_in[1];
  const float* W_out = (const float*)d_in[10];

  uint16_t* buf0 = (uint16_t*)d_ws;                                         // 32 MB (B*D*K bf16)
  uint16_t* buf1 = (uint16_t*)((char*)d_ws + (size_t)256 * 256 * 256 * 2);  // 16 MB

  k_init<<<dim3(256 * 256), dim3(256), 0, stream>>>(x, W_in, buf0);

  uint16_t* cur = buf0;
  uint16_t* nxt = buf1;
  int ncur = 256;
  for (int l = 0; l < 8; ++l) {
    const int nnext = ncur >> 1;
    const float* Wl = (const float*)d_in[2 + l];
    k_level<<<dim3(nnext), dim3(512), 0, stream>>>(cur, Wl, nxt);
    uint16_t* t = cur; cur = nxt; nxt = t;
    ncur = nnext;
  }
  k_out<<<dim3(1), dim3(256), 0, stream>>>(cur, W_out, (float*)d_out);
}
